// DynamicEdgeConv_17583596110567
// MI455X (gfx1250) — matrix operand order repaired and checked by reading, not compiled, not run on hardware
//
#include <hip/hip_runtime.h>
#include <math.h>

typedef __bf16 bf16_t;
typedef __bf16 v8bf  __attribute__((ext_vector_type(8)));
typedef __bf16 v16bf __attribute__((ext_vector_type(16)));
typedef float  v8f   __attribute__((ext_vector_type(8)));

#define FEAT   128
#define KDIM   256              // 2*FEAT
#define KPAD   264              // LDS column stride: 528B = 132 dwords == 4 mod 64 banks
#define EPW    32               // edges per wave: two 16-row WMMA tiles
#define WAVES  8
#define BLOCK  (WAVES * 32)

// ---- prep 1: W1 [256 x 128] row-major f32 -> W1T [128 x 256] row-major bf16 ----
__global__ void w1_transpose_kernel(const float* __restrict__ W1,
                                    bf16_t* __restrict__ W1T) {
    int idx = blockIdx.x * blockDim.x + threadIdx.x;   // 0 .. 32767
    if (idx < FEAT * KDIM) {
        int n = idx >> 8;
        int k = idx & 255;
        W1T[idx] = (bf16_t)W1[k * FEAT + n];
    }
}

// ---- prep 2: x f32 -> bf16 (one HBM pass; halves all gather traffic) ----
__global__ void x_to_bf16_kernel(const float* __restrict__ x,
                                 bf16_t* __restrict__ xb, int n8) {
    int i = blockIdx.x * blockDim.x + threadIdx.x;     // one 8-float chunk each
    if (i < n8) {
        const float4 a = ((const float4*)x)[2 * i];
        const float4 b = ((const float4*)x)[2 * i + 1];
        v8bf o;
        o[0] = (bf16_t)a.x; o[1] = (bf16_t)a.y; o[2] = (bf16_t)a.z; o[3] = (bf16_t)a.w;
        o[4] = (bf16_t)b.x; o[5] = (bf16_t)b.y; o[6] = (bf16_t)b.z; o[7] = (bf16_t)b.w;
        ((v8bf*)xb)[i] = o;
    }
}

// Build one 16x32 bf16 A fragment (ISA layout: per lane, elems 0-7 = K{sel*8..+8},
// elems 8-15 = K{16+sel*8..+8}) for K-chunk c of the edge this lane owns.
template <bool BF16X>
__device__ __forceinline__ v16bf a_frag(const float* __restrict__ xf,
                                        const bf16_t* __restrict__ xb,
                                        int nodeR, int nodeC, int c, int sel) {
    const int f0 = (c & 3) * 32 + sel * 8;
    const int node = (c < 4) ? nodeR : nodeC;
    if constexpr (BF16X) {
        const bf16_t* src = xb + (size_t)node * FEAT + f0;
        const v8bf lo = *(const v8bf*)(src);
        const v8bf hi = *(const v8bf*)(src + 16);
        return __builtin_shufflevector(lo, hi, 0,1,2,3,4,5,6,7,8,9,10,11,12,13,14,15);
    } else {
        const float* src = xf + (size_t)node * FEAT + f0;
        const float4 p0 = *(const float4*)(src);
        const float4 p1 = *(const float4*)(src + 4);
        const float4 q0 = *(const float4*)(src + 16);
        const float4 q1 = *(const float4*)(src + 20);
        v16bf a;
        a[0]  = (bf16_t)p0.x; a[1]  = (bf16_t)p0.y; a[2]  = (bf16_t)p0.z; a[3]  = (bf16_t)p0.w;
        a[4]  = (bf16_t)p1.x; a[5]  = (bf16_t)p1.y; a[6]  = (bf16_t)p1.z; a[7]  = (bf16_t)p1.w;
        a[8]  = (bf16_t)q0.x; a[9]  = (bf16_t)q0.y; a[10] = (bf16_t)q0.z; a[11] = (bf16_t)q0.w;
        a[12] = (bf16_t)q1.x; a[13] = (bf16_t)q1.y; a[14] = (bf16_t)q1.z; a[15] = (bf16_t)q1.w;
        return a;
    }
}

template <bool BF16X>
__global__ __launch_bounds__(BLOCK)
void edge_mlp_kernel(const float*  __restrict__ xf,
                     const bf16_t* __restrict__ xb,
                     const int*    __restrict__ edge_index,
                     const bf16_t* __restrict__ W1T,
                     const float*  __restrict__ b1,
                     const float*  __restrict__ W2,
                     const float*  __restrict__ b2,
                     float*        __restrict__ out,
                     int nEdges, int nPairs)
{
    __shared__ bf16_t ldsB[FEAT * KPAD];       // W1^T, padded columns (~66 KB)

    const int tid    = threadIdx.x;
    const int lane   = tid & 31;
    const int waveId = tid >> 5;

    // ---- stage W1^T into LDS once per block via async global->LDS DMA ----
    {
#pragma unroll
        for (int i = 0; i < (FEAT * KDIM / 8) / BLOCK; ++i) {
            const int q  = i * BLOCK + tid;    // chunk id, 8 bf16 (16B) per chunk
            const int n  = q >> 5;             // column (32 chunks per column)
            const int k8 = q & 31;
            unsigned int ldsAddr = (unsigned int)(uintptr_t)(ldsB + n * KPAD + k8 * 8);
            unsigned long long ga = (unsigned long long)(uintptr_t)(W1T + (size_t)q * 8);
            asm volatile("global_load_async_to_lds_b128 %0, %1, off"
                         :: "v"(ldsAddr), "v"(ga) : "memory");
        }
        asm volatile("s_wait_asynccnt 0x0" ::: "memory");
    }
    __syncthreads();

    const int pair = blockIdx.x * WAVES + waveId;
    if (pair >= nPairs) return;                // wave-uniform: EXEC stays all-1s

    const int eBase = pair * EPW;
    const int m     = lane & 15;               // row this lane owns within a tile
    const int sel   = lane >> 4;               // which K-half this lane holds

    int e0 = eBase + m;          if (e0 >= nEdges) e0 = nEdges - 1;
    int e1 = eBase + 16 + m;     if (e1 >= nEdges) e1 = nEdges - 1;
    const int nodeR0 = edge_index[e0];
    const int nodeC0 = edge_index[nEdges + e0];
    const int nodeR1 = edge_index[e1];
    const int nodeC1 = edge_index[nEdges + e1];

    // ---- layer 1: two 16x256 @ 256x128 matmuls sharing every B fragment ----
    v8f acc0[8] = {};
    v8f acc1[8] = {};

#pragma unroll
    for (int c = 0; c < 8; ++c) {
        const v16bf a0 = a_frag<BF16X>(xf, xb, nodeR0, nodeC0, c, sel);
        const v16bf a1 = a_frag<BF16X>(xf, xb, nodeR1, nodeC1, c, sel);
        const int kb = c * 32 + sel * 8;       // K{sel*8..+8} and K{16+sel*8..+8} per lane (ISA B layout)
        const bf16_t* bbase = ldsB + m * KPAD + kb;

        // software double-buffer the B fragments so WMMAs overlap LDS latency
        v8bf blo = *(const v8bf*)(bbase);
        v8bf bhi = *(const v8bf*)(bbase + 16);
#pragma unroll
        for (int nt = 0; nt < 8; ++nt) {
            v8bf nlo = blo, nhi = bhi;
            if (nt < 7) {
                const bf16_t* nb = bbase + (nt + 1) * 16 * KPAD;
                nlo = *(const v8bf*)(nb);
                nhi = *(const v8bf*)(nb + 16);
            }
            const v16bf bfrag = __builtin_shufflevector(
                blo, bhi, 0,1,2,3,4,5,6,7,8,9,10,11,12,13,14,15);
            acc0[nt] = __builtin_amdgcn_wmma_f32_16x16x32_bf16(
                false, a0, false, bfrag, (short)0, acc0[nt], false, false);
            acc1[nt] = __builtin_amdgcn_wmma_f32_16x16x32_bf16(
                false, a1, false, bfrag, (short)0, acc1[nt], false, false);
            blo = nlo; bhi = nhi;
        }
    }

    // ---- layer 2: bias + relu, dot with W2 across 128 channels ----
    // D layout: lane carries column n = nt*16+m; VGPR j is row sel*8+j.
    float p0[8], p1[8];
#pragma unroll
    for (int j = 0; j < 8; ++j) { p0[j] = 0.0f; p1[j] = 0.0f; }

#pragma unroll
    for (int nt = 0; nt < 8; ++nt) {
        const int n = nt * 16 + m;
        const float bias = b1[n];
        const float w2n  = W2[n];
#pragma unroll
        for (int j = 0; j < 8; ++j) {
            float h0 = acc0[nt][j] + bias;  h0 = h0 > 0.0f ? h0 : 0.0f;
            float h1 = acc1[nt][j] + bias;  h1 = h1 > 0.0f ? h1 : 0.0f;
            p0[j] += h0 * w2n;
            p1[j] += h1 * w2n;
        }
    }

    // butterfly reduce over the 16 lanes sharing the same row half
#pragma unroll
    for (int mask = 1; mask < 16; mask <<= 1) {
#pragma unroll
        for (int j = 0; j < 8; ++j) {
            p0[j] += __shfl_xor(p0[j], mask, 32);
            p1[j] += __shfl_xor(p1[j], mask, 32);
        }
    }

    if (m == 0) {
        const float bias2 = b2[0];
#pragma unroll
        for (int j = 0; j < 8; ++j) {
            const int ea = eBase + sel * 8 + j;
            const int eb = eBase + 16 + sel * 8 + j;
            if (ea < nEdges) out[ea] = 1.0f / (1.0f + __expf(-(p0[j] + bias2)));
            if (eb < nEdges) out[eb] = 1.0f / (1.0f + __expf(-(p1[j] + bias2)));
        }
    }
}

extern "C" void kernel_launch(void* const* d_in, const int* in_sizes, int n_in,
                              void* d_out, int out_size, void* d_ws, size_t ws_size,
                              hipStream_t stream) {
    const float* x   = (const float*)d_in[0];
    const int*   ei  = (const int*)  d_in[1];
    const float* W1  = (const float*)d_in[2];
    const float* b1v = (const float*)d_in[3];
    const float* W2v = (const float*)d_in[4];
    const float* b2v = (const float*)d_in[5];
    float* out = (float*)d_out;

    const int E   = in_sizes[1] / 2;                // 600000
    const int nX  = in_sizes[0];                    // 12.8M floats
    bf16_t* W1T = (bf16_t*)d_ws;                    // 64 KB
    bf16_t* XB  = (bf16_t*)((char*)d_ws + 65536);   // 25.6 MB bf16 copy of x
    const size_t needXB = 65536 + (size_t)nX * sizeof(bf16_t);

    w1_transpose_kernel<<<(FEAT * KDIM + 255) / 256, 256, 0, stream>>>(W1, W1T);

    const int nPairs  = (E + EPW - 1) / EPW;        // 18750 wave pairs
    const int nBlocks = (nPairs + WAVES - 1) / WAVES;

    if (ws_size >= needXB) {
        const int n8 = nX / 8;
        x_to_bf16_kernel<<<(n8 + 255) / 256, 256, 0, stream>>>(x, XB, n8);
        edge_mlp_kernel<true><<<nBlocks, BLOCK, 0, stream>>>(
            x, XB, ei, W1T, b1v, W2v, b2v, out, E, nPairs);
    } else {
        edge_mlp_kernel<false><<<nBlocks, BLOCK, 0, stream>>>(
            x, (const bf16_t*)nullptr, ei, W1T, b1v, W2v, b2v, out, E, nPairs);
    }
}
